// TritonEvoformerBlock_63797444215484
// MI455X (gfx1250) — compile-verified, matrix operand order checked by reading, not run on hardware
//
#include <hip/hip_runtime.h>
#include <hip/hip_bf16.h>

// ---------------------------------------------------------------------------
// CDNA5 / gfx1250 Evoformer block.
// LDS-staged batched GEMM: block computes 128x64 C; 8 waves x 4 WMMA/K-step.
// Tiles are moved global->LDS with GLOBAL_LOAD_ASYNC_TO_LDS_B128 (ASYNCcnt)
// where contiguous, f32->f16 conversion happens at fragment build.
// ---------------------------------------------------------------------------

typedef __attribute__((ext_vector_type(16))) _Float16 v16h;
typedef __attribute__((ext_vector_type(8)))  float    v8f;
typedef __attribute__((ext_vector_type(4)))  int      v4i;

#define BM 128
#define BN 64
#define BK 32
#define PKF 36   // padded LDS row stride in floats (keeps 16B alignment, spreads banks)

#ifndef __has_builtin
#define __has_builtin(x) 0
#endif

#if __has_builtin(__builtin_amdgcn_global_load_async_to_lds_b128)
#define ASYNC_LDS 1
#else
#define ASYNC_LDS 0
#endif

// 16B global -> LDS move. Async (ASYNCcnt) when available, float4 copy else.
__device__ __forceinline__ void copy16_g2l(const float* __restrict__ g,
                                           float* __restrict__ l) {
#if ASYNC_LDS
  __builtin_amdgcn_global_load_async_to_lds_b128((v4i*)g, (v4i*)l, 0, 0);
#else
  *(float4*)l = *(const float4*)g;
#endif
}

__device__ __forceinline__ void wait_async_all() {
#if ASYNC_LDS
#if __has_builtin(__builtin_amdgcn_s_wait_asynccnt)
  __builtin_amdgcn_s_wait_asynccnt(0);
#else
  asm volatile("s_wait_asynccnt 0x0" ::: "memory");
#endif
#endif
}

// ----------------------------- A / B accessors -----------------------------
// AMODE 0: row-major A[m*lda + k]
// AMODE 1: transposed A[k*lda + m]   (ax^T in outer-product-mean)
// AMODE 2: gather from outer matrix (8192x8192): m=(i*256+j), k=(d*32+e)
template<int AMODE>
__device__ __forceinline__ const float* addrA(const float* __restrict__ A, long base,
                                              int m, int k, int lda) {
  if constexpr (AMODE == 0) {
    return A + base + (long)m * lda + k;
  } else if constexpr (AMODE == 1) {
    return A + base + (long)k * lda + m;
  } else {
    const int i = m >> 8, j = m & 255;   // S = 256
    const int d = k >> 5, e = k & 31;    // OD = 32
    return A + (((long)(i * 32 + d)) << 13) + (j * 32 + e);  // ld = 8192
  }
}

// BMODE 0: row-major B[k*ldb + n]
// BMODE 1: transposed B[n*ldb + k]   (K^T for attention scores)
template<int BMODE>
__device__ __forceinline__ const float* addrB(const float* __restrict__ B, long base,
                                              int k, int n, int ldb) {
  if constexpr (BMODE == 0) return B + base + (long)k * ldb + n;
  else                      return B + base + (long)n * ldb + k;
}

// ------------------------- generic batched WMMA GEMM -----------------------
// grid = (ceil(N/64), M/128, batches); batch b -> (b1=b/binner, b2=b%binner).
// Requires M % 128 == 0, K % 32 == 0. Async-filled operands additionally
// require full-width N (true for every launch using them here).
// C = act( alpha * A@B + Res )
template<int AMODE, int BMODE>
__global__ __launch_bounds__(256)
void gemm_wmma(const float* __restrict__ A, const float* __restrict__ B,
               float* __restrict__ C, const float* __restrict__ Res,
               int M, int N, int K, int lda, int ldb, int ldc,
               long sA1, long sA2, long sB1, long sB2, long sC1, long sC2,
               int binner, float alpha, int relu)
{
  __shared__ float Asf[BM * PKF];   // [m][k] k-fast, f32
  __shared__ float Bsf[BN * PKF];   // [n][k] k-fast (B transposed in LDS), f32

  const int tid   = threadIdx.x;
  const int lane  = tid & 31;
  const int wave  = tid >> 5;
  const int mbase = blockIdx.y * BM;
  const int nbase = blockIdx.x * BN;

  const int b1 = blockIdx.z / binner, b2 = blockIdx.z % binner;
  const long offA = (long)b1 * sA1 + (long)b2 * sA2;
  const long offB = (long)b1 * sB1 + (long)b2 * sB2;
  const long offC = (long)b1 * sC1 + (long)b2 * sC2;

  const int g  = lane >> 4;       // half-wave group (ISA 7.12.2)
  const int nc = lane & 15;       // row-in-tile for A / col-in-tile for B,C

  // cooperative fill assignment
  const int am = tid >> 1;            // A row      0..127
  const int ak = (tid & 1) * 16;      // A k-offset 0/16 (16 consecutive k)
  const int bn = tid >> 2;            // B col      0..63
  const int bk = (tid & 3) * 8;       // B k-offset (8 consecutive k)
  const int bN = nbase + bn;

  v8f acc[4] = {};

  for (int k0 = 0; k0 < K; k0 += BK) {
    // Speculative prefetch of next K slab (global_prefetch_b8); unconditional
    // addresses so no exec-mask juggling.
    if (k0 + BK < K) {
      __builtin_prefetch(addrA<AMODE>(A, offA, mbase + am, k0 + BK + ak, lda), 0, 1);
      __builtin_prefetch(addrB<BMODE>(B, offB, k0 + BK + bk, bN, ldb), 0, 1);
    }

    // --- A tile: global -> LDS ---
    if constexpr (AMODE != 1) {            // contiguous along k: async 16B moves
#pragma unroll
      for (int c = 0; c < 4; ++c)
        copy16_g2l(addrA<AMODE>(A, offA, mbase + am, k0 + ak + 4 * c, lda),
                   &Asf[am * PKF + ak + 4 * c]);
    } else {                               // k-major: strided gather
#pragma unroll
      for (int t = 0; t < 16; ++t)
        Asf[am * PKF + ak + t] = *addrA<AMODE>(A, offA, mbase + am, k0 + ak + t, lda);
    }

    // --- B tile: global -> LDS (stored k-fast / transposed) ---
    if constexpr (BMODE == 1) {            // contiguous along k: async 16B moves
#pragma unroll
      for (int c = 0; c < 2; ++c)
        copy16_g2l(addrB<BMODE>(B, offB, k0 + bk + 4 * c, bN, ldb),
                   &Bsf[bn * PKF + bk + 4 * c]);
    } else {                               // n-fast in memory: strided gather + guard
#pragma unroll
      for (int t = 0; t < 8; ++t) {
        float v = 0.f;
        if (bN < N) v = *addrB<BMODE>(B, offB, k0 + bk + t, bN, ldb);
        Bsf[bn * PKF + bk + t] = v;
      }
    }

    if constexpr (AMODE != 1 || BMODE == 1) wait_async_all();
    __syncthreads();

    // --- one A fragment per wave (ISA 16-bit A layout), f32->f16 here ---
    v16h af;
#pragma unroll
    for (int j = 0; j < 8; ++j) {
      const int kk = 2 * (j & 3) + 8 * g + 16 * (j >> 2);
      af[2 * j]     = (_Float16)Asf[(wave * 16 + nc) * PKF + kk];
      af[2 * j + 1] = (_Float16)Asf[(wave * 16 + nc) * PKF + kk + 1];
    }
    // --- 4 B fragments -> 4 WMMAs, A reused ---
#pragma unroll
    for (int f = 0; f < 4; ++f) {
      v16h bf;
#pragma unroll
      for (int j = 0; j < 8; ++j) {
        const int kk = 2 * (j & 3) + 8 * g + 16 * (j >> 2);
        bf[2 * j]     = (_Float16)Bsf[(f * 16 + nc) * PKF + kk];
        bf[2 * j + 1] = (_Float16)Bsf[(f * 16 + nc) * PKF + kk + 1];
      }
      acc[f] = __builtin_amdgcn_wmma_f32_16x16x32_f16(
          /*neg_a=*/false, af, /*neg_b=*/false, bf,
          /*c_mod=*/(short)0, acc[f], /*reuse_a=*/false, /*reuse_b=*/false);
    }
    __syncthreads();
  }

  // --- epilogue: C/D layout VGPR r -> M = r + 8*(lane/16), N = lane%16 ---
#pragma unroll
  for (int f = 0; f < 4; ++f) {
    const int n = nbase + f * 16 + nc;
    if (n < N) {
#pragma unroll
      for (int r = 0; r < 8; ++r) {
        const int m = mbase + wave * 16 + r + 8 * g;
        const long idx = offC + (long)m * ldc + n;
        float vv = acc[f][r] * alpha;
        if (Res)  vv += Res[idx];
        if (relu) vv = vv > 0.f ? vv : 0.f;
        C[idx] = vv;
      }
    }
  }
}

// ------------------------------- LayerNorm ---------------------------------
__global__ __launch_bounds__(32)
void ln_kernel(const float* __restrict__ x, const float* __restrict__ w,
               float* __restrict__ y, int D)
{
  const long row = blockIdx.x;
  const int lane = threadIdx.x;
  const float* xr = x + row * D;
  float s = 0.f, ss = 0.f;
  for (int i = lane; i < D; i += 32) { float v = xr[i]; s += v; ss += v * v; }
  for (int o = 16; o > 0; o >>= 1) {
    s  += __shfl_xor(s,  o, 32);
    ss += __shfl_xor(ss, o, 32);
  }
  const float mean = s / (float)D;
  const float var  = ss / (float)D - mean * mean;
  const float rstd = rsqrtf(var + 1e-5f);
  float* yr = y + row * D;
  for (int i = lane; i < D; i += 32) yr[i] = (xr[i] - mean) * rstd * w[i];
}

// -------------------------------- Softmax ----------------------------------
// One wave per row of length L. Optional pair-bias: rows enumerate (n,h,i),
// bias element for column j is bias[(i*Ss + j)*Hh + h].
__global__ __launch_bounds__(32)
void softmax_kernel(float* __restrict__ s, int L,
                    const float* __restrict__ bias, int Hh, int Ss)
{
  const long row = blockIdx.x;
  const int lane = threadIdx.x;
  float* sr = s + row * L;

  int h = 0, i = 0;
  if (bias) {
    i = (int)(row % Ss);
    h = (int)((row / Ss) % Hh);
  }

  float mx = -3.4e38f;
  for (int j = lane; j < L; j += 32) {
    float v = sr[j];
    if (bias) v += bias[((long)i * Ss + j) * Hh + h];
    mx = fmaxf(mx, v);
  }
  for (int o = 16; o > 0; o >>= 1) mx = fmaxf(mx, __shfl_xor(mx, o, 32));

  float sum = 0.f;
  for (int j = lane; j < L; j += 32) {
    float v = sr[j];
    if (bias) v += bias[((long)i * Ss + j) * Hh + h];
    v = __expf(v - mx);
    sr[j] = v;
    sum += v;
  }
  for (int o = 16; o > 0; o >>= 1) sum += __shfl_xor(sum, o, 32);
  const float inv = 1.f / sum;
  for (int j = lane; j < L; j += 32) sr[j] *= inv;
}

// ------------------------------- host launch -------------------------------
extern "C" void kernel_launch(void* const* d_in, const int* in_sizes, int n_in,
                              void* d_out, int out_size, void* d_ws, size_t ws_size,
                              hipStream_t stream)
{
  (void)in_sizes; (void)n_in; (void)out_size; (void)ws_size;

  const int Nn = 128, S = 256, D = 256, PD = 128, OD = 32, H = 8, HD = 32;
  const int MI = 4 * D;   // 1024
  const int PI = 4 * PD;  // 512
  const float scale = 1.0f / sqrtf((float)HD);

  const float* msa_in     = (const float*)d_in[0];
  const float* pair_in    = (const float*)d_in[1];
  const float* ln_row_w   = (const float*)d_in[2];
  const float* row_wq     = (const float*)d_in[3];
  const float* row_wk     = (const float*)d_in[4];
  const float* row_wv     = (const float*)d_in[5];
  const float* row_wo     = (const float*)d_in[6];
  const float* row_bias_w = (const float*)d_in[7];
  const float* ln_col_w   = (const float*)d_in[8];
  const float* col_wq     = (const float*)d_in[9];
  const float* col_wk     = (const float*)d_in[10];
  const float* col_wv     = (const float*)d_in[11];
  const float* col_wo     = (const float*)d_in[12];
  const float* ln_mt_w    = (const float*)d_in[13];
  const float* mt_w1      = (const float*)d_in[14];
  const float* mt_w2      = (const float*)d_in[15];
  const float* opm_ln_w   = (const float*)d_in[16];
  const float* opm_proj_w = (const float*)d_in[17];
  const float* opm_out_w  = (const float*)d_in[18];
  const float* ln_pt_w    = (const float*)d_in[19];
  const float* pt_w1      = (const float*)d_in[20];
  const float* pt_w2      = (const float*)d_in[21];

  const size_t MSA_E  = (size_t)Nn * S * D;   // 8,388,608
  const size_t PAIR_E = (size_t)S * S * PD;   // 8,388,608

  float* ws = (float*)d_ws;
  size_t off = 0;
  auto take = [&](size_t n) { float* r = ws + off; off += n; return r; };
  float* xln   = take(MSA_E);                 // reused for pair LN too
  float* q     = take(MSA_E);
  float* k     = take(MSA_E);
  float* v     = take(MSA_E);
  float* o     = take(MSA_E);
  float* biasb = take((size_t)S * S * H);     // (S*S, H)
  float* msa1  = take(MSA_E);
  float* msa2  = take(MSA_E);
  float* ax    = take((size_t)Nn * S * OD);   // (N*S, OD)
  float* pair1 = take(PAIR_E);
  float* big   = take((size_t)67108864);      // scores / hidden / outer (reused)

  float* msa_out  = (float*)d_out;
  float* pair_out = msa_out + MSA_E;

  const dim3 blk(256);
  const long Z = 0;

  // ======================= MSA row attention (+pair bias) ==================
  ln_kernel<<<Nn * S, 32, 0, stream>>>(msa_in, ln_row_w, xln, D);
  gemm_wmma<0,0><<<dim3(4, 256, 1), blk, 0, stream>>>(xln, row_wq, q, nullptr,
      Nn*S, D, D, D, D, D, Z,Z,Z,Z,Z,Z, 1, 1.f, 0);
  gemm_wmma<0,0><<<dim3(4, 256, 1), blk, 0, stream>>>(xln, row_wk, k, nullptr,
      Nn*S, D, D, D, D, D, Z,Z,Z,Z,Z,Z, 1, 1.f, 0);
  gemm_wmma<0,0><<<dim3(4, 256, 1), blk, 0, stream>>>(xln, row_wv, v, nullptr,
      Nn*S, D, D, D, D, D, Z,Z,Z,Z,Z,Z, 1, 1.f, 0);
  // pair bias: (S*S, PD) @ (PD, H) -> (S*S, H)   (N=8 handled by lane guards)
  gemm_wmma<0,0><<<dim3(1, 512, 1), blk, 0, stream>>>(pair_in, row_bias_w, biasb, nullptr,
      S*S, H, PD, PD, H, H, Z,Z,Z,Z,Z,Z, 1, 1.f, 0);
  // scores[n,h] = scale * Q_nh @ K_nh^T   batches (n,h)
  gemm_wmma<0,1><<<dim3(4, 2, Nn*H), blk, 0, stream>>>(q, k, big, nullptr,
      S, S, HD, D, D, S,
      (long)S*D, (long)HD, (long)S*D, (long)HD, (long)H*S*S, (long)S*S,
      H, scale, 0);
  softmax_kernel<<<Nn * H * S, 32, 0, stream>>>(big, S, biasb, H, S);
  // O[n,h] = A @ V_nh
  gemm_wmma<0,0><<<dim3(1, 2, Nn*H), blk, 0, stream>>>(big, v, o, nullptr,
      S, HD, S, S, D, D,
      (long)H*S*S, (long)S*S, (long)S*D, (long)HD, (long)S*D, (long)HD,
      H, 1.f, 0);
  // msa1 = msa + O @ row_wo
  gemm_wmma<0,0><<<dim3(4, 256, 1), blk, 0, stream>>>(o, row_wo, msa1, msa_in,
      Nn*S, D, D, D, D, D, Z,Z,Z,Z,Z,Z, 1, 1.f, 0);

  // =========================== MSA column attention ========================
  ln_kernel<<<Nn * S, 32, 0, stream>>>(msa1, ln_col_w, xln, D);
  gemm_wmma<0,0><<<dim3(4, 256, 1), blk, 0, stream>>>(xln, col_wq, q, nullptr,
      Nn*S, D, D, D, D, D, Z,Z,Z,Z,Z,Z, 1, 1.f, 0);
  gemm_wmma<0,0><<<dim3(4, 256, 1), blk, 0, stream>>>(xln, col_wk, k, nullptr,
      Nn*S, D, D, D, D, D, Z,Z,Z,Z,Z,Z, 1, 1.f, 0);
  gemm_wmma<0,0><<<dim3(4, 256, 1), blk, 0, stream>>>(xln, col_wv, v, nullptr,
      Nn*S, D, D, D, D, D, Z,Z,Z,Z,Z,Z, 1, 1.f, 0);
  // scores[i,h] over (n,m): batches (i,h); rows stride S*D
  gemm_wmma<0,1><<<dim3(2, 1, S*H), blk, 0, stream>>>(q, k, big, nullptr,
      Nn, Nn, HD, S*D, S*D, Nn,
      (long)D, (long)HD, (long)D, (long)HD, (long)H*Nn*Nn, (long)Nn*Nn,
      H, scale, 0);
  softmax_kernel<<<S * H * Nn, 32, 0, stream>>>(big, Nn, nullptr, 1, 1);
  gemm_wmma<0,0><<<dim3(1, 1, S*H), blk, 0, stream>>>(big, v, o, nullptr,
      Nn, HD, Nn, Nn, S*D, S*D,
      (long)H*Nn*Nn, (long)Nn*Nn, (long)D, (long)HD, (long)D, (long)HD,
      H, 1.f, 0);
  // msa2 = msa1 + O @ col_wo
  gemm_wmma<0,0><<<dim3(4, 256, 1), blk, 0, stream>>>(o, col_wo, msa2, msa1,
      Nn*S, D, D, D, D, D, Z,Z,Z,Z,Z,Z, 1, 1.f, 0);

  // ============================== MSA transition ===========================
  ln_kernel<<<Nn * S, 32, 0, stream>>>(msa2, ln_mt_w, xln, D);
  gemm_wmma<0,0><<<dim3(16, 256, 1), blk, 0, stream>>>(xln, mt_w1, big, nullptr,
      Nn*S, MI, D, D, MI, MI, Z,Z,Z,Z,Z,Z, 1, 1.f, 1 /*relu*/);
  gemm_wmma<0,0><<<dim3(4, 256, 1), blk, 0, stream>>>(big, mt_w2, msa_out, msa2,
      Nn*S, D, MI, MI, D, D, Z,Z,Z,Z,Z,Z, 1, 1.f, 0);

  // ============================ Outer product mean =========================
  ln_kernel<<<Nn * S, 32, 0, stream>>>(msa_out, opm_ln_w, xln, D);
  gemm_wmma<0,0><<<dim3(1, 256, 1), blk, 0, stream>>>(xln, opm_proj_w, ax, nullptr,
      Nn*S, OD, D, D, OD, OD, Z,Z,Z,Z,Z,Z, 1, 1.f, 0);
  // outer (S*OD x S*OD) = (1/N) * ax^T @ ax,  ax viewed as (N x S*OD)
  gemm_wmma<1,0><<<dim3(128, 64, 1), blk, 0, stream>>>(ax, ax, big, nullptr,
      S*OD, S*OD, Nn, S*OD, S*OD, S*OD, Z,Z,Z,Z,Z,Z, 1, 1.f / (float)Nn, 0);
  // pair1 = pair + reshape(outer) @ opm_out_w   (gathered-A mode)
  gemm_wmma<2,0><<<dim3(2, 512, 1), blk, 0, stream>>>(big, opm_out_w, pair1, pair_in,
      S*S, PD, OD*OD, S*OD, PD, PD, Z,Z,Z,Z,Z,Z, 1, 1.f, 0);

  // ============================== Pair transition ==========================
  ln_kernel<<<S * S, 32, 0, stream>>>(pair1, ln_pt_w, xln, PD);
  gemm_wmma<0,0><<<dim3(8, 512, 1), blk, 0, stream>>>(xln, pt_w1, big, nullptr,
      S*S, PI, PD, PD, PI, PI, Z,Z,Z,Z,Z,Z, 1, 1.f, 1 /*relu*/);
  gemm_wmma<0,0><<<dim3(2, 512, 1), blk, 0, stream>>>(big, pt_w2, pair_out, pair1,
      S*S, PD, PI, PI, PD, PD, Z,Z,Z,Z,Z,Z, 1, 1.f, 0);
}
